// LFQQuantizerEMA_KMeans_21895743275552
// MI455X (gfx1250) — matrix operand order repaired and checked by reading, not compiled, not run on hardware
//
#include <hip/hip_runtime.h>

// ---------------------------------------------------------------------------
// VQ-VAE EMA quantizer for MI455X (gfx1250, wave32, WMMA bf16 + TDM path)
// ---------------------------------------------------------------------------

typedef __attribute__((ext_vector_type(16))) __bf16 v16bf;
typedef __attribute__((ext_vector_type(8)))  __bf16 v8bf;
typedef __attribute__((ext_vector_type(8)))  float  v8f;
typedef __attribute__((ext_vector_type(4)))  unsigned int u32x4;
typedef __attribute__((ext_vector_type(8)))  int    i32x8;
typedef __attribute__((ext_vector_type(4)))  int    i32x4;

#define DECAY        0.99f
#define OMD          0.01f          // 1 - DECAY
#define EPSV         1e-5f
#define DEAD_THRESH  5.0f

#define BROWS   131072
#define DDIM    64
#define KCODES  1024
#define CBSTRIDE 72                 // bf16 elems per LDS codebook row (36-bank stagger)

#define CBBYTES  (KCODES*CBSTRIDE*2)   // 147456
#define C2BYTES  (KCODES*4)            // 4096
#define STG_ROWS 128                   // rows per TDM chunk
#define STGBYTES (2*STG_ROWS*DDIM*4)   // 65536 (double buffer)

// workspace layout (float offsets)
#define WS_CS    0
#define WS_ES    (KCODES)
#define WS_NCB   (WS_ES + KCODES*DDIM)
#define WS_NC2   (WS_NCB + KCODES*DDIM)
#define WS_DEAD  (WS_NC2 + KCODES)
#define WS_ANY   (WS_DEAD + KCODES)

#define BIGF 3.0e38f

#define HAVE_TDM __has_builtin(__builtin_amdgcn_tensor_load_to_lds)

__device__ __forceinline__ v16bf pack_a(float4 p0, float4 p1, float4 p2, float4 p3) {
    v16bf v;
    v[0]=(__bf16)p0.x;  v[1]=(__bf16)p0.y;  v[2]=(__bf16)p0.z;  v[3]=(__bf16)p0.w;
    v[4]=(__bf16)p1.x;  v[5]=(__bf16)p1.y;  v[6]=(__bf16)p1.z;  v[7]=(__bf16)p1.w;
    v[8]=(__bf16)p2.x;  v[9]=(__bf16)p2.y;  v[10]=(__bf16)p2.z; v[11]=(__bf16)p2.w;
    v[12]=(__bf16)p3.x; v[13]=(__bf16)p3.y; v[14]=(__bf16)p3.z; v[15]=(__bf16)p3.w;
    return v;
}

__device__ __forceinline__ v16bf cat8(v8bf lo, v8bf hi) {
    return __builtin_shufflevector(lo, hi, 0,1,2,3,4,5,6,7,8,9,10,11,12,13,14,15);
}

__device__ __forceinline__ v16bf load_a_pair(const float* zr, int abase, int doff) {
    float4 p0 = *(const float4*)(zr + abase + doff);
    float4 p1 = *(const float4*)(zr + abase + doff + 4);
    float4 p2 = *(const float4*)(zr + abase + doff + 8);
    float4 p3 = *(const float4*)(zr + abase + doff + 12);
    return pack_a(p0, p1, p2, p3);
}

// Stage one f32 row (64 elems) into LDS as bf16, return sum of squares (f32).
__device__ __forceinline__ float stage_row_bf16(const float* __restrict__ src,
                                                __bf16* __restrict__ dst) {
    float c2 = 0.f;
#pragma unroll
    for (int c = 0; c < 8; ++c) {
        float4 f0 = *(const float4*)(src + c*8);
        float4 f1 = *(const float4*)(src + c*8 + 4);
        c2 += f0.x*f0.x + f0.y*f0.y + f0.z*f0.z + f0.w*f0.w
            + f1.x*f1.x + f1.y*f1.y + f1.z*f1.z + f1.w*f1.w;
        v8bf pk;
        pk[0]=(__bf16)f0.x; pk[1]=(__bf16)f0.y; pk[2]=(__bf16)f0.z; pk[3]=(__bf16)f0.w;
        pk[4]=(__bf16)f1.x; pk[5]=(__bf16)f1.y; pk[6]=(__bf16)f1.z; pk[7]=(__bf16)f1.w;
        *(v8bf*)(dst + c*8) = pk;
    }
    return c2;
}

#if HAVE_TDM
__device__ __forceinline__ unsigned lds_offset_of(const void* p) {
    return (unsigned)(size_t)(__attribute__((address_space(3))) const char*)p;
}

// Issue one TDM 2D tile load: `rows` x 64 f32, contiguous, global -> LDS.
// Descriptor per CDNA5 ISA ch.8: D# group0 + group1 (groups 2/3 zero, 2D).
__device__ __forceinline__ void tdm_load_rows(const float* gsrc, unsigned ldsByteOff,
                                              int rows) {
    unsigned long long ga = (unsigned long long)(size_t)gsrc;
    u32x4 g0;
    g0[0] = 1u;                                               // count=1, user mode
    g0[1] = ldsByteOff;                                       // lds_addr
    g0[2] = (unsigned)(ga & 0xFFFFFFFFu);                     // global_addr[31:0]
    g0[3] = ((unsigned)(ga >> 32) & 0x01FFFFFFu) | 0x80000000u; // addr[56:32] | type=2
    i32x8 g1;
    g1[0] = (int)(2u << 16);              // data_size=4B; mask=0; no pad/iter
    g1[1] = (int)((unsigned)DDIM << 16);  // tensor_dim0[15:0] @ bits 63:48
    g1[2] = (int)((unsigned)rows << 16);  // tensor_dim1[15:0] @ bits 95:80
    g1[3] = (int)((unsigned)DDIM << 16);  // tile_dim0 @ bits 127:112
    g1[4] = rows;                         // tile_dim1 @ bits 143:128
    g1[5] = DDIM;                         // tensor_dim0_stride[31:0] @ bits 191:160
    g1[6] = 0;
    g1[7] = 0;
    i32x4 z4 = {0, 0, 0, 0};
    i32x8 z8 = {0, 0, 0, 0, 0, 0, 0, 0};
    __builtin_amdgcn_tensor_load_to_lds(g0, g1, z4, z4, z8, 0);
}
#endif

// ---------------------------------------------------------------------------
// Kernel 1: nearest-code assignment. 512 threads = 16 waves; each wave owns
// two 16-row tiles (32 rows); block covers 512 rows; full codebook in LDS
// (bf16), staged via TDM (double-buffered f32 chunks) when available.
// Main loop is software-pipelined: argmin(j-1) runs between the B-loads and
// WMMAs of tile j, hiding DS latency and the WMMA->VALU RAW hazard.
// ---------------------------------------------------------------------------
__global__ __launch_bounds__(512) void vq_assign_kernel(
    const float* __restrict__ z_e, const float* __restrict__ codebook,
    float* __restrict__ zq, float* __restrict__ idx_out,
    float* __restrict__ ws_cs, float* __restrict__ ws_es)
{
    extern __shared__ char smem[];
    __bf16* ldsCB  = (__bf16*)smem;
    float*  ldsC2  = (float*)(smem + CBBYTES);
    float*  ldsStg = (float*)(smem + CBBYTES + C2BYTES);
    int*    ldsIdx = (int*)(smem + CBBYTES + C2BYTES + STGBYTES);

    const int tid  = threadIdx.x;
    const int lane = tid & 31;
    const int wave = tid >> 5;
    const int hi   = lane >> 4;
    const int l16  = lane & 15;

    // ---------------- codebook staging ----------------
#if HAVE_TDM
    {
        const unsigned stgOff = lds_offset_of(ldsStg);
        if (wave == 0) tdm_load_rows(codebook, stgOff, STG_ROWS);
        for (int c = 0; c < 8; ++c) {
            if (wave == 0) {
                if (c + 1 < 8) {
                    tdm_load_rows(codebook + (size_t)(c+1)*STG_ROWS*DDIM,
                                  stgOff + (unsigned)(((c+1)&1)*STG_ROWS*DDIM*4),
                                  STG_ROWS);
                    __builtin_amdgcn_s_wait_tensorcnt(1);   // chunk c done
                } else {
                    __builtin_amdgcn_s_wait_tensorcnt(0);   // last chunk done
                }
            }
            __syncthreads();                       // chunk c resident in LDS
            // convert chunk c: 4 threads per row (16 elems each)
            {
                const int row  = tid >> 2;          // 0..127
                const int part = tid & 3;
                const float* srow = ldsStg + (size_t)(c&1)*STG_ROWS*DDIM + (size_t)row*DDIM;
                __bf16* drow = ldsCB + (size_t)(c*STG_ROWS + row)*CBSTRIDE + part*16;
                const float* s16 = srow + part*16;
#pragma unroll
                for (int q = 0; q < 2; ++q) {
                    float4 f0 = *(const float4*)(s16 + q*8);
                    float4 f1 = *(const float4*)(s16 + q*8 + 4);
                    v8bf pk;
                    pk[0]=(__bf16)f0.x; pk[1]=(__bf16)f0.y; pk[2]=(__bf16)f0.z; pk[3]=(__bf16)f0.w;
                    pk[4]=(__bf16)f1.x; pk[5]=(__bf16)f1.y; pk[6]=(__bf16)f1.z; pk[7]=(__bf16)f1.w;
                    *(v8bf*)(drow + q*8) = pk;
                }
                if (part == 0) {
                    float c2 = 0.f;
#pragma unroll
                    for (int d = 0; d < DDIM; ++d) c2 += srow[d]*srow[d];
                    ldsC2[c*STG_ROWS + row] = c2;
                }
            }
            __syncthreads();                       // conversion done before buf reuse
        }
    }
#else
    for (int k = tid; k < KCODES; k += 512)
        ldsC2[k] = stage_row_bf16(codebook + (size_t)k*DDIM, ldsCB + (size_t)k*CBSTRIDE);
    __syncthreads();
#endif

    // ---------------- main WMMA argmin loop (pipelined) ----------------
    const int row0 = blockIdx.x * 512 + wave * 32;
    const float* zr0 = z_e + (size_t)(row0 + l16) * DDIM;
    const float* zr1 = z_e + (size_t)(row0 + 16 + l16) * DDIM;

    const int abase = hi * 16;
    v16bf a00 = load_a_pair(zr0, abase, 0);
    v16bf a01 = load_a_pair(zr0, abase, 32);
    v16bf a10 = load_a_pair(zr1, abase, 0);
    v16bf a11 = load_a_pair(zr1, abase, 32);

    float mv[16]; int mi[16];
#pragma unroll
    for (int r = 0; r < 16; ++r) { mv[r] = BIGF; mi[r] = 0; }

    const int kb2 = hi * 16;
    const __bf16* brBase = ldsCB + (size_t)l16 * CBSTRIDE + kb2;

    // prologue: tile 0
    v8f accP0 = {}, accP1 = {};
    {
        const __bf16* br = brBase;
        v16bf b0 = cat8(*(const v8bf*)(br),      *(const v8bf*)(br + 8));
        v16bf b1 = cat8(*(const v8bf*)(br + 32), *(const v8bf*)(br + 40));
        accP0 = __builtin_amdgcn_wmma_f32_16x16x32_bf16(false, a00, false, b0,
                                                        (short)0, accP0, false, false);
        accP0 = __builtin_amdgcn_wmma_f32_16x16x32_bf16(false, a01, false, b1,
                                                        (short)0, accP0, false, false);
        accP1 = __builtin_amdgcn_wmma_f32_16x16x32_bf16(false, a10, false, b0,
                                                        (short)0, accP1, false, false);
        accP1 = __builtin_amdgcn_wmma_f32_16x16x32_bf16(false, a11, false, b1,
                                                        (short)0, accP1, false, false);
    }
    float c2p = ldsC2[l16];

    for (int j = 1; j < 64; ++j) {
        // B-loads for tile j (in flight during argmin below)
        const __bf16* br = brBase + (size_t)j * 16 * CBSTRIDE;
        v16bf b0 = cat8(*(const v8bf*)(br),      *(const v8bf*)(br + 8));
        v16bf b1 = cat8(*(const v8bf*)(br + 32), *(const v8bf*)(br + 40));
        const float c2n = ldsC2[j*16 + l16];

        // argmin for tile j-1 (accP from previous iteration's WMMAs; only the B-loads sit in between)
        const int code = (j-1)*16 + l16;
#pragma unroll
        for (int r = 0; r < 8; ++r) {
            float s0 = c2p - 2.0f * accP0[r];
            if (s0 < mv[r])     { mv[r]     = s0; mi[r]     = code; }
            float s1 = c2p - 2.0f * accP1[r];
            if (s1 < mv[8 + r]) { mv[8 + r] = s1; mi[8 + r] = code; }
        }

        // WMMAs for tile j
        v8f acc0 = {}, acc1 = {};
        acc0 = __builtin_amdgcn_wmma_f32_16x16x32_bf16(false, a00, false, b0,
                                                       (short)0, acc0, false, false);
        acc0 = __builtin_amdgcn_wmma_f32_16x16x32_bf16(false, a01, false, b1,
                                                       (short)0, acc0, false, false);
        acc1 = __builtin_amdgcn_wmma_f32_16x16x32_bf16(false, a10, false, b0,
                                                       (short)0, acc1, false, false);
        acc1 = __builtin_amdgcn_wmma_f32_16x16x32_bf16(false, a11, false, b1,
                                                       (short)0, acc1, false, false);
        accP0 = acc0; accP1 = acc1; c2p = c2n;
    }
    // tail: argmin for tile 63
    {
        const int code = 63*16 + l16;
#pragma unroll
        for (int r = 0; r < 8; ++r) {
            float s0 = c2p - 2.0f * accP0[r];
            if (s0 < mv[r])     { mv[r]     = s0; mi[r]     = code; }
            float s1 = c2p - 2.0f * accP1[r];
            if (s1 < mv[8 + r]) { mv[8 + r] = s1; mi[8 + r] = code; }
        }
    }

    // cross-lane argmin within each 16-lane group (lexicographic, argmin-first)
#pragma unroll
    for (int off = 1; off < 16; off <<= 1) {
#pragma unroll
        for (int r = 0; r < 16; ++r) {
            float ov = __shfl_xor(mv[r], off, 32);
            int   oi = __shfl_xor(mi[r], off, 32);
            if (ov < mv[r] || (ov == mv[r] && oi < mi[r])) { mv[r] = ov; mi[r] = oi; }
        }
    }

    if (l16 == 0) {
#pragma unroll
        for (int t = 0; t < 2; ++t) {
            const int base = wave*32 + t*16 + hi*8;
#pragma unroll
            for (int r = 0; r < 8; ++r) ldsIdx[base + r] = mi[t*8 + r];
        }
    }
    __syncthreads();

    // ---------------- outputs ----------------
#pragma unroll
    for (int t = 0; t < 2; ++t) {
        const int row  = row0 + t*16 + l16;
        const int cidx = ldsIdx[wave*32 + t*16 + l16];
        if (hi == 0) {
            idx_out[row] = (float)cidx;
            atomicAdd(ws_cs + cidx, 1.0f);
        }
        const float* csrc = codebook + (size_t)cidx*DDIM + hi*32;
        float* zdst = zq + (size_t)row*DDIM + hi*32;
#pragma unroll
        for (int c = 0; c < 8; ++c) ((float4*)zdst)[c] = ((const float4*)csrc)[c];

        const float* zsrc = z_e + (size_t)row*DDIM + hi*32;
        float* es = ws_es + (size_t)cidx*DDIM + hi*32;
#pragma unroll
        for (int i = 0; i < 32; ++i) atomicAdd(es + i, zsrc[i]);
    }
}

// ---------------------------------------------------------------------------
// Kernel 2: EMA update, normalization, dead flags. One block of 1024 threads.
// ---------------------------------------------------------------------------
__global__ __launch_bounds__(1024) void vq_finalize_kernel(
    const float* __restrict__ ema_cs, const float* __restrict__ ema_cb,
    const float* __restrict__ usage, float* __restrict__ ws)
{
    __shared__ float sred[1024];
    const int k = threadIdx.x;

    const float cs    = ws[WS_CS + k];
    const float necs  = DECAY * ema_cs[k] + OMD * cs;
    const float u     = usage[k] + cs;
    const float deadf = (u < DEAD_THRESH) ? 1.0f : 0.0f;

    sred[k] = necs; __syncthreads();
#pragma unroll
    for (int s = 512; s > 0; s >>= 1) { if (k < s) sred[k] += sred[k + s]; __syncthreads(); }
    const float n = sred[0];
    __syncthreads();

    sred[k] = 1.0f - deadf; __syncthreads();
#pragma unroll
    for (int s = 512; s > 0; s >>= 1) { if (k < s) sred[k] += sred[k + s]; __syncthreads(); }
    const float aliveCnt = sred[0];

    const float norm = (necs + EPSV) / (n + (float)KCODES * EPSV);
    const float inv  = 1.0f / norm;

    const float* er  = ema_cb + (size_t)k*DDIM;
    const float* esr = ws + WS_ES  + (size_t)k*DDIM;
    float*       out = ws + WS_NCB + (size_t)k*DDIM;
    float nc2 = 0.f;
#pragma unroll 8
    for (int d = 0; d < DDIM; ++d) {
        const float v = (DECAY * er[d] + OMD * esr[d]) * inv;
        out[d] = v;
        nc2 += v * v;
    }
    ws[WS_NC2 + k]  = nc2;
    ws[WS_DEAD + k] = deadf;
    if (k == 0) ws[WS_ANY] = (aliveCnt > 0.5f) ? 1.0f : 0.0f;
}

// ---------------------------------------------------------------------------
// Kernel 3: dead-code replacement via K x K WMMA distances (alive columns).
// ---------------------------------------------------------------------------
__global__ __launch_bounds__(256) void vq_dead_replace_kernel(
    const float* __restrict__ ws, float* __restrict__ fcb)
{
    extern __shared__ char smem[];
    __bf16* ldsCB   = (__bf16*)smem;
    float*  ldsC2   = (float*)(smem + CBBYTES);
    float*  ldsDead = ldsC2 + KCODES;
    int*    ldsIdx  = (int*)(ldsDead + KCODES);

    const int tid  = threadIdx.x;
    const int lane = tid & 31;
    const int wave = tid >> 5;
    const int hi   = lane >> 4;
    const int l16  = lane & 15;

    const float* ncb = ws + WS_NCB;
    for (int k = tid; k < KCODES; k += 256) {
        (void)stage_row_bf16(ncb + (size_t)k*DDIM, ldsCB + (size_t)k*CBSTRIDE);
        ldsC2[k]   = ws[WS_NC2 + k];
        ldsDead[k] = ws[WS_DEAD + k];
    }
    __syncthreads();

    const float anyAlive = ws[WS_ANY];
    const int row0 = blockIdx.x * 128 + wave * 16;
    const float* rr = ncb + (size_t)(row0 + l16) * DDIM;

    const int abase = hi * 16;
    v16bf a0 = load_a_pair(rr, abase, 0);
    v16bf a1 = load_a_pair(rr, abase, 32);

    float mv[8]; int mi[8];
#pragma unroll
    for (int r = 0; r < 8; ++r) { mv[r] = BIGF; mi[r] = 0; }

    const int kb2 = hi * 16;
    for (int j = 0; j < 64; ++j) {
        const int code = j*16 + l16;
        const __bf16* br = ldsCB + (size_t)code * CBSTRIDE + kb2;
        v16bf b0 = cat8(*(const v8bf*)(br),      *(const v8bf*)(br + 8));
        v16bf b1 = cat8(*(const v8bf*)(br + 32), *(const v8bf*)(br + 40));
        v8f acc = {};
        acc = __builtin_amdgcn_wmma_f32_16x16x32_bf16(false, a0, false, b0,
                                                      (short)0, acc, false, false);
        acc = __builtin_amdgcn_wmma_f32_16x16x32_bf16(false, a1, false, b1,
                                                      (short)0, acc, false, false);
        const float c2eff = (ldsDead[code] != 0.f) ? BIGF : ldsC2[code];
#pragma unroll
        for (int r = 0; r < 8; ++r) {
            float s = c2eff - 2.0f * acc[r];
            if (s < mv[r]) { mv[r] = s; mi[r] = code; }
        }
    }

#pragma unroll
    for (int off = 1; off < 16; off <<= 1) {
#pragma unroll
        for (int r = 0; r < 8; ++r) {
            float ov = __shfl_xor(mv[r], off, 32);
            int   oi = __shfl_xor(mi[r], off, 32);
            if (ov < mv[r] || (ov == mv[r] && oi < mi[r])) { mv[r] = ov; mi[r] = oi; }
        }
    }

    if (l16 == 0) {
        const int base = wave*16 + hi*8;
#pragma unroll
        for (int r = 0; r < 8; ++r) ldsIdx[base + r] = mi[r];
    }
    __syncthreads();

    {
        const int row = row0 + l16;
        const int nearest = ldsIdx[wave*16 + l16];
        const float deadr = ldsDead[row];
        const int src = (deadr != 0.f && anyAlive != 0.f) ? nearest : row;
        const float* s = ncb + (size_t)src*DDIM + hi*32;
        float* d = fcb + (size_t)row*DDIM + hi*32;
#pragma unroll
        for (int c = 0; c < 8; ++c) ((float4*)d)[c] = ((const float4*)s)[c];
    }
}

// ---------------------------------------------------------------------------
extern "C" void kernel_launch(void* const* d_in, const int* in_sizes, int n_in,
                              void* d_out, int out_size, void* d_ws, size_t ws_size,
                              hipStream_t stream) {
    const float* z_e      = (const float*)d_in[0];
    const float* codebook = (const float*)d_in[1];
    const float* ema_cs   = (const float*)d_in[2];
    const float* ema_cb   = (const float*)d_in[3];
    const float* usage    = (const float*)d_in[4];

    float* out     = (float*)d_out;
    float* zq      = out;
    float* idx_out = out + (size_t)BROWS * DDIM;
    float* fcb     = idx_out + BROWS;
    float* ws      = (float*)d_ws;

    (void)hipMemsetAsync(ws, 0, (size_t)(WS_ES + KCODES*DDIM) * sizeof(float), stream);

    const size_t shmem1 = (size_t)CBBYTES + C2BYTES + STGBYTES + 512*4;
    vq_assign_kernel<<<BROWS/512, 512, shmem1, stream>>>(
        z_e, codebook, zq, idx_out, ws + WS_CS, ws + WS_ES);

    vq_finalize_kernel<<<1, 1024, 0, stream>>>(ema_cs, ema_cb, usage, ws);

    const size_t shmem3 = (size_t)CBBYTES + C2BYTES + KCODES*4 + 128*4;
    vq_dead_replace_kernel<<<KCODES/128, 256, shmem3, stream>>>(ws, fcb);
}
